// AdaTripletMiner_65996467470339
// MI455X (gfx1250) — matrix operand order repaired and checked by reading, not compiled, not run on hardware
//
#include <hip/hip_runtime.h>

typedef __attribute__((ext_vector_type(16))) _Float16 v16h;
typedef __attribute__((ext_vector_type(8)))  _Float16 v8h;
typedef __attribute__((ext_vector_type(8)))  float    v8f;
typedef __attribute__((ext_vector_type(4)))  float    f4;

#define NN   384
#define KK   1024
#define NCLS 32

// ---------------- Kernel 1: row L2-normalize, emit f16 ----------------
__global__ __launch_bounds__(256)
void k_normalize(const float* __restrict__ logits, _Float16* __restrict__ xh) {
  int row = blockIdx.x;
  const float* r = logits + (size_t)row * KK;
  float s = 0.f;
  for (int c = threadIdx.x; c < KK; c += 256) { float v = r[c]; s += v * v; }
  #pragma unroll
  for (int off = 16; off > 0; off >>= 1) s += __shfl_down(s, off);
  __shared__ float red[8];
  __shared__ float rinv_s;
  if ((threadIdx.x & 31) == 0) red[threadIdx.x >> 5] = s;
  __syncthreads();
  if (threadIdx.x == 0) {
    float t = 0.f;
    for (int i = 0; i < 8; ++i) t += red[i];
    rinv_s = rsqrtf(t);
  }
  __syncthreads();
  float rinv = rinv_s;
  for (int c = threadIdx.x; c < KK; c += 256)
    xh[(size_t)row * KK + c] = (_Float16)(r[c] * rinv);
}

// ------------- Kernel 2: mat = -(x @ x^T) via v_wmma_f32_16x16x32_f16 -------------
// One wave per 16x16 output tile. A: 16-bit 16x32 layout (lane lo=M, hi selects
// K half-groups -> two contiguous 8-half (16B) chunks per lane). B: 32x16 layout
// (lane lo=N column = row of x, hi selects K half-groups -> same two 16B chunks as A).
__global__ __launch_bounds__(32)
void k_gemm(const _Float16* __restrict__ xh, float* __restrict__ mat) {
  int lane = threadIdx.x;
  int lo = lane & 15, hi = lane >> 4;
  const _Float16* arow = xh + (size_t)(blockIdx.y * 16 + lo) * KK;
  const _Float16* brow = xh + (size_t)(blockIdx.x * 16 + lo) * KK;
  v8f c = {0.f, 0.f, 0.f, 0.f, 0.f, 0.f, 0.f, 0.f};
  for (int k0 = 0; k0 < KK; k0 += 32) {
    v8h a0 = *(const v8h*)(arow + k0 + 8 * hi);
    v8h a1 = *(const v8h*)(arow + k0 + 16 + 8 * hi);
    v8h b0 = *(const v8h*)(brow + k0 + 8 * hi);
    v8h b1 = *(const v8h*)(brow + k0 + 16 + 8 * hi);
    v16h a, b;
    #pragma unroll
    for (int i = 0; i < 8; ++i) {
      a[i] = a0[i]; a[8 + i] = a1[i];
      b[i] = b0[i]; b[8 + i] = b1[i];
    }
    c = __builtin_amdgcn_wmma_f32_16x16x32_f16(false, a, false, b,
                                               (short)0, c, false, false);
  }
  // D layout: VGPR r -> M = r + 8*hi, N = lo
  int m0 = blockIdx.y * 16 + 8 * hi;
  int n  = blockIdx.x * 16 + lo;
  #pragma unroll
  for (int r = 0; r < 8; ++r)
    mat[(size_t)(m0 + r) * NN + n] = -c[r];
}

// ---------------- Kernel 3: class counts (int atomics, deterministic) ----------------
__global__ __launch_bounds__(NN)
void k_counts(const int* __restrict__ labels, int* __restrict__ counts) {
  int t = threadIdx.x;
  if (t < NCLS) counts[t] = 0;
  __syncthreads();
  atomicAdd(&counts[labels[t]], 1);
}

// ---------------- Kernel 4: per-anchor partial sums (O(N^2) total) ----------------
__global__ __launch_bounds__(128)
void k_anchor(const float* __restrict__ mat, const int* __restrict__ labels,
              const int* __restrict__ counts,
              float* __restrict__ tripPart, float* __restrict__ tripCnt,
              float* __restrict__ anPart, float* __restrict__ anCnt) {
  int a = blockIdx.x;
  int la = labels[a];
  int cl = counts[la];
  float nPos = (float)(cl - 1);
  float nNeg = (float)(NN - cl);
  bool hpa = cl >= 2;
  float sPos = 0.f, sNeg = 0.f, sAn = 0.f, cAn = 0.f;
  for (int j = threadIdx.x; j < NN; j += 128) {
    int lj = labels[j];
    float m = mat[(size_t)a * NN + j];
    if (lj == la) {
      if (j != a) sPos += m;
    } else {
      sNeg += m;
      if (j > a && (hpa || counts[lj] >= 2)) { sAn += m; cAn += 1.f; }
    }
  }
  #pragma unroll
  for (int off = 16; off > 0; off >>= 1) {
    sPos += __shfl_down(sPos, off);
    sNeg += __shfl_down(sNeg, off);
    sAn  += __shfl_down(sAn,  off);
    cAn  += __shfl_down(cAn,  off);
  }
  __shared__ float sh[16];
  int wid = threadIdx.x >> 5;
  if ((threadIdx.x & 31) == 0) {
    sh[wid * 4 + 0] = sPos; sh[wid * 4 + 1] = sNeg;
    sh[wid * 4 + 2] = sAn;  sh[wid * 4 + 3] = cAn;
  }
  __syncthreads();
  if (threadIdx.x == 0) {
    float tp = 0, tn = 0, ta = 0, tc = 0;
    for (int w = 0; w < 4; ++w) {
      tp += sh[w * 4 + 0]; tn += sh[w * 4 + 1];
      ta += sh[w * 4 + 2]; tc += sh[w * 4 + 3];
    }
    // sum over triplets of delta = nPos * sumNeg - nNeg * sumPos
    tripPart[a] = nPos * tn - nNeg * tp;
    tripCnt[a]  = nPos * nNeg;
    anPart[a] = ta;
    anCnt[a]  = tc;
  }
}

// ---------------- Kernel 5: finalize scalars eps & beta ----------------
__global__ __launch_bounds__(32)
void k_finalize(const float* __restrict__ tripPart, const float* __restrict__ tripCnt,
                const float* __restrict__ anPart, const float* __restrict__ anCnt,
                float* __restrict__ scalars, float* __restrict__ outTail) {
  int t = threadIdx.x;
  float s0 = 0, s1 = 0, s2 = 0, s3 = 0;
  for (int i = t; i < NN; i += 32) {
    s0 += tripPart[i]; s1 += tripCnt[i];
    s2 += anPart[i];   s3 += anCnt[i];
  }
  #pragma unroll
  for (int off = 16; off > 0; off >>= 1) {
    s0 += __shfl_down(s0, off); s1 += __shfl_down(s1, off);
    s2 += __shfl_down(s2, off); s3 += __shfl_down(s3, off);
  }
  if (t == 0) {
    float eps = (s0 / fmaxf(s1, 1.0f)) * 0.5f;        // / K_DELTA
    eps = fminf(fmaxf(eps, 0.0f), 0.5f);
    float an_mean = s2 / fmaxf(s3, 1.0f);
    float beta = 1.0f + (-an_mean - 1.0f) * 0.5f;     // / K_AN
    scalars[0] = eps; scalars[1] = beta;
    outTail[0] = eps; outTail[1] = beta;
  }
}

// ---------------- Kernel 6: an_pairs [N,N] ----------------
__global__ __launch_bounds__(96)
void k_anpairs(const float* __restrict__ mat, const int* __restrict__ labels,
               const int* __restrict__ counts, const float* __restrict__ scalars,
               float* __restrict__ out) {
  int a = blockIdx.x;
  int la = labels[a];
  bool hpa = counts[la] >= 2;
  float beta = scalars[1];
  int n0 = threadIdx.x * 4;
  f4 r;
  #pragma unroll
  for (int i = 0; i < 4; ++i) {
    int n = n0 + i;
    int ln = labels[n];
    bool v = (n > a) && (ln != la) && (hpa || counts[ln] >= 2) &&
             (-mat[(size_t)a * NN + n] >= beta);
    r[i] = v ? 1.0f : 0.0f;
  }
  *(f4*)(out + (size_t)a * NN + n0) = r;
}

// ---------------- Kernel 7: cond [N,N,N] — the 227 MB write (BW-bound) ----------------
__global__ __launch_bounds__(96)
void k_cond(const float* __restrict__ mat, const int* __restrict__ labels,
            const float* __restrict__ scalars, float* __restrict__ out) {
  int p = blockIdx.x, a = blockIdx.y;
  int la = labels[a];
  size_t base = ((size_t)a * NN + p) * NN;
  int n0 = threadIdx.x * 4;
  f4 r = {0.f, 0.f, 0.f, 0.f};
  if ((labels[p] == la) && (a != p)) {            // sames[a,p]
    float eps = scalars[0];
    float map = mat[(size_t)a * NN + p];
    f4 man = *(const f4*)(mat + (size_t)a * NN + n0);
    #pragma unroll
    for (int i = 0; i < 4; ++i) {
      int n = n0 + i;
      float d = man[i] - map;                     // delta[a,p,n]
      bool v = (labels[n] != la) && (d <= eps) && (d > 0.0f);
      r[i] = v ? 1.0f : 0.0f;
    }
  }
  __builtin_nontemporal_store(r, (f4*)(out + base + n0));
}

extern "C" void kernel_launch(void* const* d_in, const int* in_sizes, int n_in,
                              void* d_out, int out_size, void* d_ws, size_t ws_size,
                              hipStream_t stream) {
  const float* logits = (const float*)d_in[0];
  const int*   labels = (const int*)d_in[1];
  float* out = (float*)d_out;

  // workspace layout (bytes)
  char* ws = (char*)d_ws;
  _Float16* xh  = (_Float16*)ws;                              // 384*1024*2 = 786432
  float*    mat = (float*)(ws + 786432);                      // 384*384*4  = 589824
  int*   counts = (int*)(ws + 786432 + 589824);               // 128
  float* tripPart = (float*)(ws + 1376384);                   // 384*4 each
  float* tripCnt  = tripPart + NN;
  float* anPart   = tripCnt + NN;
  float* anCnt    = anPart + NN;
  float* scalars  = anCnt + NN;                               // [eps, beta]

  const size_t N3 = (size_t)NN * NN * NN;
  const size_t N2 = (size_t)NN * NN;

  k_normalize<<<NN, 256, 0, stream>>>(logits, xh);
  k_gemm<<<dim3(NN / 16, NN / 16), 32, 0, stream>>>(xh, mat);
  k_counts<<<1, NN, 0, stream>>>(labels, counts);
  k_anchor<<<NN, 128, 0, stream>>>(mat, labels, counts,
                                   tripPart, tripCnt, anPart, anCnt);
  k_finalize<<<1, 32, 0, stream>>>(tripPart, tripCnt, anPart, anCnt,
                                   scalars, out + N3 + N2);
  k_anpairs<<<NN, 96, 0, stream>>>(mat, labels, counts, scalars, out + N3);
  k_cond<<<dim3(NN, NN), 96, 0, stream>>>(mat, labels, scalars, out);
}
